// Temporal_Multimodal_Decoder_28355374088297
// MI455X (gfx1250) — matrix operand order repaired and checked by reading, not compiled, not run on hardware
//
#include <hip/hip_runtime.h>
#include <hip/hip_bf16.h>

#define OBS    20
#define PRED   30
#define MODES  6
#define HDIM   256
#define BATCH  8192
#define TILE_B 64          // batch rows per block
#define KCH    10          // K chunks of 32 halves: 2 for padded x (64), 8 for h (256)
#define NCT    16          // 16-wide column tiles per gate (256/16)
#define FRAG_HALVES 512    // one 32x16 f16 B fragment = 512 halves = 1KB
#define NFRAGS (64 * KCH)  // 64 gate-column tiles (4 gates * 16) x 10 k-chunks

#define LOG2E 1.4426950408889634f

typedef __attribute__((ext_vector_type(16))) _Float16 v16h;
typedef __attribute__((ext_vector_type(8)))  _Float16 v8h;
typedef __attribute__((ext_vector_type(8)))  float    v8f;

union Frag16 {
    v16h v;
    v8h  h8[2];
    _Float16 h[16];
};

// Fast TRANS-pipe nonlinearities: v_exp_f32 + v_rcp_f32 co-execute with the
// matrix pipe on CDNA5 (XDL WMMA is tracked as TRANS for scheduling).
__device__ __forceinline__ float fast_sigmoid(float x) {
    return __builtin_amdgcn_rcpf(1.0f + __builtin_amdgcn_exp2f(-x * LOG2E));
}
__device__ __forceinline__ float fast_tanh(float x) {
    return 1.0f - 2.0f * __builtin_amdgcn_rcpf(__builtin_amdgcn_exp2f(x * (2.0f * LOG2E)) + 1.0f);
}

// ---------------------------------------------------------------------------
// Pack [W_ih | W_hh] (f32) into f16 B-matrix fragments: fragment (ct,kc)
// covers gate cols ct*16..+15, K chunk kc. Lane l owns col ct*16+(l&15),
// K = ((l&16)?8:0)+j (+8 for j>=8), stored contiguously -> 32B per lane in the GEMM.
// ---------------------------------------------------------------------------
__global__ __launch_bounds__(256) void pack_kernel(const float* __restrict__ W_ih,
                                                   const float* __restrict__ W_hh,
                                                   _Float16* __restrict__ packed) {
    int gid = blockIdx.x * blockDim.x + threadIdx.x;
    if (gid >= NFRAGS * 32) return;
    int lane = gid & 31;
    int frag = gid >> 5;
    int kc   = frag % KCH;
    int ct   = frag / KCH;                 // 0..63 = gate*16 + nt
    int col  = ct * 16 + (lane & 15);      // gate column 0..1023
    int kbase = (lane & 16) ? 8 : 0;
    _Float16* dst = packed + (size_t)frag * FRAG_HALVES + lane * 16;
#pragma unroll
    for (int j = 0; j < 16; ++j) {
        int K = kbase + j + ((j >= 8) ? 8 : 0);
        float v;
        if (kc < 2) {                      // padded x region: K-space 0..63, valid 0..39
            int k = kc * 32 + K;
            v = (k < 40) ? W_ih[col * 40 + k] : 0.0f;
        } else {                           // h region: K-space 0..255
            int k = (kc - 2) * 32 + K;
            v = W_hh[col * 256 + k];
        }
        dst[j] = (_Float16)v;
    }
}

// ---------------------------------------------------------------------------
// Main LSTM decoder: one block = one (mode, 64-row batch tile); 4 waves of 32,
// each wave owns 16 rows. c kept in VGPRs in WMMA C/D layout; h kept in LDS
// as f16 in A-feed position. 640 v_wmma_f32_16x16x32_f16 per wave per step.
// The weight pointer is laundered through inline asm each step so LICM cannot
// hoist the (t-invariant) B-fragment loads out of the step loop -- that
// hoisting is what blew the 1024-VGPR file and spilled in prior builds.
// ---------------------------------------------------------------------------
__global__ __launch_bounds__(128) void decoder_kernel(
    const float* __restrict__ traj_rel,   // (OBS, B, 2)
    const float* __restrict__ state_h,    // (1, B, H)
    const float* __restrict__ c0,         // (M, B, H)
    const float* __restrict__ W_se,       // (40, 40)
    const float* __restrict__ b_se,       // (40)
    const float* __restrict__ b_ih,       // (1024)
    const float* __restrict__ b_hh,       // (1024)
    const float* __restrict__ W_pred,     // (M, 2, H)
    const float* __restrict__ b_pred,     // (M, 2)
    const _Float16* __restrict__ wpk,     // packed gate weights (NFRAGS * 512 halves)
    float* __restrict__ out_pred)         // (B, M, PRED, 2)
{
    __shared__ _Float16 xh[TILE_B][320];      // [0..39]=x  [40..63]=0 pad  [64..319]=h   (40KB)
    __shared__ float    win[TILE_B][OBS][2];  // rolling rel-displacement window          (10KB)
    __shared__ float    wse[40][40];          //                                          (6.4KB)
    __shared__ float    bse[40];
    __shared__ float    bias[1024];           // b_ih + b_hh                              (4KB)
    __shared__ float    wp[2][HDIM];          // this mode's prediction head              (2KB)
    __shared__ float    bp[2];

    const int tid     = threadIdx.x;
    const int lane    = tid & 31;
    const int wave    = tid >> 5;
    const int mode    = blockIdx.x % MODES;
    const int rowBase = (blockIdx.x / MODES) * TILE_B;

    // ---------------- init LDS ----------------
    for (int i = tid; i < 1600; i += 128) wse[i / 40][i % 40] = W_se[i];
    for (int i = tid; i < 40;   i += 128) bse[i] = b_se[i];
    for (int i = tid; i < 1024; i += 128) bias[i] = b_ih[i] + b_hh[i];
    for (int i = tid; i < 512;  i += 128) wp[i >> 8][i & 255] = W_pred[(mode * 2 + (i >> 8)) * HDIM + (i & 255)];
    if (tid < 2) bp[tid] = b_pred[mode * 2 + tid];
    for (int i = tid; i < TILE_B * OBS * 2; i += 128) {
        int row = i / 40, rem = i % 40, ts = rem >> 1, d = rem & 1;
        win[row][ts][d] = traj_rel[((size_t)ts * BATCH + rowBase + row) * 2 + d];
    }
    for (int i = tid; i < TILE_B * 24; i += 128) {     // zero pad cols 40..63 once
        xh[i / 24][40 + (i % 24)] = (_Float16)0.0f;
    }
    for (int i = tid; i < TILE_B * HDIM; i += 128) {   // h0 -> f16 in A-feed position
        int row = i >> 8, k = i & 255;
        xh[row][64 + k] = (_Float16)state_h[(size_t)(rowBase + row) * HDIM + k];
    }

    // ---------------- c0 -> registers in WMMA C/D lane layout ----------------
    // D tile layout: VGPR r, lanes 0-15 -> M=r, N=lane; lanes 16-31 -> M=r+8, N=lane-16.
    float c_reg[NCT][8];
    {
        const float* cb = c0 + (size_t)mode * BATCH * HDIM;
        const int moff = (lane >= 16) ? 8 : 0;
        const int n    = lane & 15;
#pragma unroll
        for (int nt = 0; nt < NCT; ++nt)
#pragma unroll
            for (int r = 0; r < 8; ++r) {
                int grow = rowBase + wave * 16 + moff + r;
                c_reg[nt][r] = cb[(size_t)grow * HDIM + nt * 16 + n];
            }
    }
    __syncthreads();

    const int aRow = wave * 16 + (lane & 15);
    const int koff = (lane >= 16) ? 8 : 0;    // A-fragment K sub-offset per ISA layout

#pragma unroll 1
    for (int t = 0; t < PRED; ++t) {
        // Launder the weight pointer: compiler must treat it as fresh each
        // step, so B-fragment loads stay inside the loop (L2-resident stream)
        // instead of being hoisted into (and spilling) the register file.
        const _Float16* wstep = wpk;
        asm volatile("" : "+s"(wstep));

        // ---- 1) spatial embedding x = dropout(leaky_relu(win @ Wse^T + bse)) ----
        {
            const int xrow = tid & 63;
            const int cpar = tid >> 6;        // column parity (0/1)
            float wv[40];
#pragma unroll
            for (int j = 0; j < OBS; ++j) {
                int slot = (t + j) % OBS;     // oldest sample sits at slot t%OBS
                wv[j * 2]     = win[xrow][slot][0];
                wv[j * 2 + 1] = win[xrow][slot][1];
            }
#pragma unroll 2
            for (int cc = 0; cc < 20; ++cc) {
                int col = cpar + cc * 2;
                float s = bse[col];
#pragma unroll
                for (int k = 0; k < 40; ++k) s += wv[k] * wse[col][k];
                s = (s > 0.0f) ? s : 0.01f * s;                    // leaky relu
                unsigned u = (((unsigned)(mode * PRED + t) * BATCH + (rowBase + xrow)) * 64u + col) * 2654435761u;
                u ^= u >> 15; u *= 0x27d4eb2du; u ^= u >> 13;      // dropout hash
                s = ((u & 0xFFFFFFu) < 12582912u) ? s * (1.0f / 0.75f) : 0.0f;
                xh[xrow][col] = (_Float16)s;
            }
        }
        __syncthreads();

        // ---- 2) build A fragments [x(pad64) | h] for this wave's 16 rows ----
        Frag16 a[KCH];
#pragma unroll
        for (int kc = 0; kc < KCH; ++kc) {
            const _Float16* rp = &xh[aRow][kc * 32 + koff];
            a[kc].h8[0] = *(const v8h*)(rp);        // K = koff + 0..7    (ds_load_b128)
            a[kc].h8[1] = *(const v8h*)(rp + 16);   // K = koff + 16..23  (ds_load_b128)
        }
        __builtin_amdgcn_sched_barrier(0);

        // ---- 3) gates = A @ Wt : 16 col-tiles x 4 gates x 10 k-chunks of WMMA ----
#pragma unroll
        for (int nt = 0; nt < NCT; ++nt) {
            const _Float16* wtile = wstep + (size_t)nt * KCH * FRAG_HALVES;
            __builtin_prefetch(wtile + KCH * FRAG_HALVES, 0, 0);    // global_prefetch_b8
            v8f acc[4];
#pragma unroll
            for (int g = 0; g < 4; ++g) {
                v8f c = {};
                const _Float16* wb = wtile + (size_t)g * NCT * KCH * FRAG_HALVES;
#pragma unroll
                for (int kc = 0; kc < KCH; ++kc) {
                    Frag16 bf;
                    bf.v = *((const v16h*)(wb + kc * FRAG_HALVES) + lane);
                    c = __builtin_amdgcn_wmma_f32_16x16x32_f16(
                            false, a[kc].v, false, bf.v, (short)0, c, false, false);
                }
                acc[g] = c;
            }
            // ---- 4) fused LSTM cell update (lane-local, D layout; TRANS-pipe) ----
            const int n  = lane & 15;
            const float bi = bias[0 * 256 + nt * 16 + n];
            const float bf = bias[1 * 256 + nt * 16 + n];
            const float bg = bias[2 * 256 + nt * 16 + n];
            const float bo = bias[3 * 256 + nt * 16 + n];
            const int mrow0 = wave * 16 + ((lane >= 16) ? 8 : 0);
#pragma unroll
            for (int r = 0; r < 8; ++r) {
                float ig = fast_sigmoid(acc[0][r] + bi);
                float fg = fast_sigmoid(acc[1][r] + bf);
                float gg = fast_tanh(acc[2][r] + bg);
                float og = fast_sigmoid(acc[3][r] + bo);
                float cv = fg * c_reg[nt][r] + ig * gg;
                c_reg[nt][r] = cv;
                float hv = og * fast_tanh(cv);
                xh[mrow0 + r][64 + nt * 16 + n] = (_Float16)hv;   // h back to A-feed slot
            }
            // Fence memory ops + WMMA per tile (VALU/SALU/TRANS may cross):
            // bounds live set without serializing the gate math.
            __builtin_amdgcn_sched_barrier(0x406);
        }

        // ---- 5) prediction head: rel = h @ Wp^T + bp (own rows; LDS in-order) ----
        {
            const int lrow = wave * 16 + (lane >> 1);
            const int d    = lane & 1;
            float s = bp[d];
#pragma unroll 8
            for (int k = 0; k < HDIM; ++k) s += (float)xh[lrow][64 + k] * wp[d][k];
            const int grow = rowBase + lrow;
            out_pred[(((size_t)grow * MODES + mode) * PRED + t) * 2 + d] = s;
            win[lrow][t % OBS][d] = s;   // overwrite oldest window slot
        }
        __syncthreads();
    }
}

// ---------------------------------------------------------------------------
// Confidence PredictionNet (residual 60->60->60 MLP + softmax over 6 modes).
// ---------------------------------------------------------------------------
__global__ __launch_bounds__(256) void conf_kernel(
    const float* __restrict__ pred,                 // (B, M, PRED, 2)
    const float* __restrict__ w1, const float* __restrict__ b1,
    const float* __restrict__ w2, const float* __restrict__ b2,
    const float* __restrict__ wfc, const float* __restrict__ bfc,
    float* __restrict__ conf)                       // (B, M)
{
    __shared__ float sw1[60][60];
    __shared__ float sw2[60][60];
    __shared__ float sb1[60], sb2[60], swf[60];
    for (int i = threadIdx.x; i < 3600; i += 256) {
        sw1[i / 60][i % 60] = w1[i];
        sw2[i / 60][i % 60] = w2[i];
    }
    for (int i = threadIdx.x; i < 60; i += 256) {
        sb1[i] = b1[i]; sb2[i] = b2[i]; swf[i] = wfc[i];
    }
    __syncthreads();

    int b = blockIdx.x * 256 + threadIdx.x;
    if (b >= BATCH) return;
    const float bfcv = bfc[0];
    float s[MODES];
    for (int m = 0; m < MODES; ++m) {
        float x[60], y1[60];
        const float* px = pred + ((size_t)b * MODES + m) * PRED * 2;
        for (int k = 0; k < 60; ++k) x[k] = px[k];
        for (int j = 0; j < 60; ++j) {
            float a = sb1[j];
            for (int k = 0; k < 60; ++k) a += sw1[j][k] * x[k];
            y1[j] = (a > 0.0f) ? a : 0.0f;
        }
        float sc = bfcv;
        for (int j = 0; j < 60; ++j) {
            float a = sb2[j] + x[j];
            for (int k = 0; k < 60; ++k) a += sw2[j][k] * y1[k];
            a = (a > 0.0f) ? a : 0.0f;
            sc += swf[j] * a;
        }
        s[m] = sc;
    }
    float mx = s[0];
    for (int m = 1; m < MODES; ++m) mx = fmaxf(mx, s[m]);
    float den = 0.0f;
    for (int m = 0; m < MODES; ++m) {
        s[m] = __builtin_amdgcn_exp2f((s[m] - mx) * LOG2E);
        den += s[m];
    }
    float inv = __builtin_amdgcn_rcpf(den);
    for (int m = 0; m < MODES; ++m) conf[(size_t)b * MODES + m] = s[m] * inv;
}

// ---------------------------------------------------------------------------
extern "C" void kernel_launch(void* const* d_in, const int* in_sizes, int n_in,
                              void* d_out, int out_size, void* d_ws, size_t ws_size,
                              hipStream_t stream) {
    const float* traj_rel = (const float*)d_in[1];
    const float* state_h  = (const float*)d_in[2];
    const float* c0       = (const float*)d_in[3];
    const float* W_se     = (const float*)d_in[4];
    const float* b_se     = (const float*)d_in[5];
    const float* W_ih     = (const float*)d_in[6];
    const float* W_hh     = (const float*)d_in[7];
    const float* b_ih     = (const float*)d_in[8];
    const float* b_hh     = (const float*)d_in[9];
    const float* W_pred   = (const float*)d_in[10];
    const float* b_pred   = (const float*)d_in[11];
    const float* pn_w1    = (const float*)d_in[12];
    const float* pn_b1    = (const float*)d_in[13];
    const float* pn_w2    = (const float*)d_in[14];
    const float* pn_b2    = (const float*)d_in[15];
    const float* pn_wfc   = (const float*)d_in[16];
    const float* pn_bfc   = (const float*)d_in[17];

    float* out  = (float*)d_out;
    float* conf = out + (size_t)BATCH * MODES * PRED * 2;
    _Float16* packed = (_Float16*)d_ws;              // 640 KB of fragment-swizzled weights

    hipLaunchKernelGGL(pack_kernel, dim3((NFRAGS * 32 + 255) / 256), dim3(256), 0, stream,
                       W_ih, W_hh, packed);
    hipLaunchKernelGGL(decoder_kernel, dim3(MODES * (BATCH / TILE_B)), dim3(128), 0, stream,
                       traj_rel, state_h, c0, W_se, b_se, b_ih, b_hh,
                       W_pred, b_pred, packed, out);
    hipLaunchKernelGGL(conf_kernel, dim3(BATCH / 256), dim3(256), 0, stream,
                       out, pn_w1, pn_b1, pn_w2, pn_b2, pn_wfc, pn_bfc, conf);
}
